// CrossAttention_77695958385468
// MI455X (gfx1250) — matrix operand order repaired and checked by reading, not compiled, not run on hardware
//
#include <hip/hip_runtime.h>
#include <hip/hip_bf16.h>
#include <stdint.h>

typedef __bf16 bf16_t;
typedef __attribute__((ext_vector_type(16))) __bf16 v16bf;
typedef __attribute__((ext_vector_type(8)))  float  v8f;

#define H_     8
#define DH_    64
#define N_     4096
#define DQ_    320
#define DC_    768
#define L_     77
#define LP_    96      // context length padded to 3 K-steps of 32
#define C_     4
#define INNER_ 512
#define ROWS_  8192    // 2*N
#define SCALE_ 0.125f  // DH^-0.5

// ---------- helpers ----------

__device__ inline bf16_t f2bf(float f) {
  unsigned u = __builtin_bit_cast(unsigned, f);
  unsigned r = (u + 0x7FFFu + ((u >> 16) & 1u)) >> 16;   // RNE
  unsigned short s = (unsigned short)r;
  return __builtin_bit_cast(bf16_t, s);
}

__device__ inline v8f vzero8() {
  v8f z;
#pragma unroll
  for (int i = 0; i < 8; ++i) z[i] = 0.0f;
  return z;
}

__device__ inline v8f wmma_bf16(v16bf a, v16bf b, v8f c) {
  // D = A(16x32 bf16) * B(32x16 bf16) + C(f32)
  return __builtin_amdgcn_wmma_f32_16x16x32_bf16(false, a, false, b, (short)0, c, false, false);
}

// A fragment 16x32 from row-major bf16 source; lane: M = l%16, h = l/16,
// element i -> K = k0 + 16*(i>=8) + 8*h + (i&7)   (two contiguous 16B chunks)
__device__ inline v16bf load_a(const bf16_t* __restrict__ src, int ld, int row0, int k0) {
  int lane = threadIdx.x & 31;
  int m = lane & 15, h = lane >> 4;
  const bf16_t* p = src + (size_t)(row0 + m) * ld + k0 + 8 * h;
  v16bf a;
#pragma unroll
  for (int i = 0; i < 8; ++i) a[i] = p[i];
#pragma unroll
  for (int i = 0; i < 8; ++i) a[8 + i] = p[16 + i];
  return a;
}

// A fragment from fp32 source with row guard (zero-padded rows >= nrows)
__device__ inline v16bf load_a_f32(const float* __restrict__ src, int ld, int nrows,
                                   int row0, int k0) {
  int lane = threadIdx.x & 31;
  int m = lane & 15, h = lane >> 4;
  int row = row0 + m;
  v16bf a;
  if (row < nrows) {
    const float* p = src + (size_t)row * ld + k0 + 8 * h;
#pragma unroll
    for (int i = 0; i < 8; ++i) a[i] = f2bf(p[i]);
#pragma unroll
    for (int i = 0; i < 8; ++i) a[8 + i] = f2bf(p[16 + i]);
  } else {
#pragma unroll
    for (int i = 0; i < 16; ++i) a[i] = f2bf(0.0f);
  }
  return a;
}

// B fragment 32x16 from "transposed" storage: element (k,n) = bt[n*ld + k];
// lane: N = l%16, element i -> K = k0 + 16*(i>=8) + 8*(l/16) + (i&7)  (two contiguous 16B chunks)
__device__ inline v16bf load_b(const bf16_t* __restrict__ bt, int ld, int k0, int n0) {
  int lane = threadIdx.x & 31;
  int n = n0 + (lane & 15), h = lane >> 4;
  const bf16_t* p = bt + (size_t)n * ld + k0 + 8 * h;
  v16bf b;
#pragma unroll
  for (int i = 0; i < 8; ++i) b[i] = p[i];
#pragma unroll
  for (int i = 0; i < 8; ++i) b[8 + i] = p[16 + i];
  return b;
}

// ---------- kernel 1: convert inputs, transpose weights, reset max ----------

__global__ void prep_kernel(const float* __restrict__ x,
                            const float* __restrict__ Wq, const float* __restrict__ Wk,
                            const float* __restrict__ Wv, const float* __restrict__ Wo,
                            bf16_t* __restrict__ xbf, bf16_t* __restrict__ WqT,
                            bf16_t* __restrict__ WkT, bf16_t* __restrict__ WvT,
                            bf16_t* __restrict__ WoT, unsigned int* __restrict__ maxbuf) {
  size_t tid = (size_t)blockIdx.x * blockDim.x + threadIdx.x;
  size_t stride = (size_t)gridDim.x * blockDim.x;
  if (tid == 0) *maxbuf = 0x007FFFFFu;  // ordered-uint encoding of -inf
  for (size_t i = tid; i < (size_t)ROWS_ * DQ_; i += stride) xbf[i] = f2bf(x[i]);
  for (size_t i = tid; i < (size_t)INNER_ * DQ_; i += stride) {
    size_t n = i / DQ_, k = i % DQ_;
    WqT[i] = f2bf(Wq[k * INNER_ + n]);
  }
  for (size_t i = tid; i < (size_t)INNER_ * DC_; i += stride) {
    size_t n = i / DC_, k = i % DC_;
    WkT[i] = f2bf(Wk[k * INNER_ + n]);
    WvT[i] = f2bf(Wv[k * INNER_ + n]);
  }
  for (size_t i = tid; i < (size_t)DQ_ * INNER_; i += stride) {
    size_t n = i / INNER_, k = i % INNER_;
    WoT[i] = f2bf(Wo[k * DQ_ + n]);
  }
}

// ---------- kernel 2: q = x @ Wq   (8192 x 512, K=320) ----------

__global__ void __launch_bounds__(256) proj_q_kernel(const bf16_t* __restrict__ xbf,
                                                     const bf16_t* __restrict__ WqT,
                                                     bf16_t* __restrict__ qbf) {
  int wave = threadIdx.x >> 5, lane = threadIdx.x & 31;
  int wid = blockIdx.x * 8 + wave;   // 0..4095
  int mtile = wid >> 3;              // 0..511
  int ng = wid & 7;                  // 4 N-tiles each
  int row0 = mtile * 16;
  int n0 = ng * 64;
  const int KS = DQ_ / 32;           // 10 (even)
  v8f acc[4];
#pragma unroll
  for (int j = 0; j < 4; ++j) acc[j] = vzero8();
  // ping-pong double buffering (2x unroll): no register rotation, loads for
  // one parity are in flight while WMMAs consume the other parity
  v16bf a0 = load_a(xbf, DQ_, row0, 0);
  v16bf b0[4];
#pragma unroll
  for (int j = 0; j < 4; ++j) b0[j] = load_b(WqT, DQ_, 0, n0 + j * 16);
  for (int ks = 0; ks < KS; ks += 2) {
    v16bf a1 = load_a(xbf, DQ_, row0, (ks + 1) * 32);
    v16bf b1[4];
#pragma unroll
    for (int j = 0; j < 4; ++j) b1[j] = load_b(WqT, DQ_, (ks + 1) * 32, n0 + j * 16);
#pragma unroll
    for (int j = 0; j < 4; ++j) acc[j] = wmma_bf16(a0, b0[j], acc[j]);
    if (ks + 2 < KS) {
      a0 = load_a(xbf, DQ_, row0, (ks + 2) * 32);
#pragma unroll
      for (int j = 0; j < 4; ++j) b0[j] = load_b(WqT, DQ_, (ks + 2) * 32, n0 + j * 16);
    }
#pragma unroll
    for (int j = 0; j < 4; ++j) acc[j] = wmma_bf16(a1, b1[j], acc[j]);
  }
  int m = lane & 15, hh = lane >> 4;
#pragma unroll
  for (int j = 0; j < 4; ++j)
#pragma unroll
    for (int r = 0; r < 8; ++r)
      qbf[(size_t)(row0 + r + 8 * hh) * INNER_ + n0 + j * 16 + m] = f2bf(acc[j][r]);
}

// ---------- kernel 3: K/V projections of contexts ----------
// kbuf slots [5][96][512] row-major:  0=k_uc, 1+c=k_c[c]
// vtbuf slots [5][512][96] transposed: 0=v_uc, 1+c=v_c[c]

__global__ void __launch_bounds__(256) proj_ctx_kernel(const float* __restrict__ uc,
                                                       const float* __restrict__ ck,
                                                       const float* __restrict__ cv,
                                                       const bf16_t* __restrict__ WkT,
                                                       const bf16_t* __restrict__ WvT,
                                                       bf16_t* __restrict__ kbuf,
                                                       bf16_t* __restrict__ vtbuf) {
  int wave = threadIdx.x >> 5, lane = threadIdx.x & 31;
  int wid = blockIdx.x * 8 + wave;   // 0..399
  int job = wid / 40;                // 0..9
  int rem = wid % 40;
  int mtile = rem / 8;               // 0..4 (rows 0..79)
  int ng = rem % 8;                  // 4 N-tiles each
  int n0 = ng * 64;
  const float* src; const bf16_t* wt; bf16_t* dst; bool ktype;
  if (job == 0)      { src = uc; wt = WkT; dst = kbuf;  ktype = true;  }
  else if (job == 1) { src = uc; wt = WvT; dst = vtbuf; ktype = false; }
  else if (job <= 5) { int c = job - 2; src = ck + (size_t)c * L_ * DC_; wt = WkT;
                       dst = kbuf  + (size_t)(1 + c) * LP_ * INNER_; ktype = true;  }
  else               { int c = job - 6; src = cv + (size_t)c * L_ * DC_; wt = WvT;
                       dst = vtbuf + (size_t)(1 + c) * LP_ * INNER_; ktype = false; }
  const int KS = DC_ / 32;           // 24 (even)
  v8f acc[4];
#pragma unroll
  for (int j = 0; j < 4; ++j) acc[j] = vzero8();
  v16bf a0 = load_a_f32(src, DC_, L_, mtile * 16, 0);
  v16bf b0[4];
#pragma unroll
  for (int j = 0; j < 4; ++j) b0[j] = load_b(wt, DC_, 0, n0 + j * 16);
  for (int ks = 0; ks < KS; ks += 2) {
    v16bf a1 = load_a_f32(src, DC_, L_, mtile * 16, (ks + 1) * 32);
    v16bf b1[4];
#pragma unroll
    for (int j = 0; j < 4; ++j) b1[j] = load_b(wt, DC_, (ks + 1) * 32, n0 + j * 16);
#pragma unroll
    for (int j = 0; j < 4; ++j) acc[j] = wmma_bf16(a0, b0[j], acc[j]);
    if (ks + 2 < KS) {
      a0 = load_a_f32(src, DC_, L_, mtile * 16, (ks + 2) * 32);
#pragma unroll
      for (int j = 0; j < 4; ++j) b0[j] = load_b(wt, DC_, (ks + 2) * 32, n0 + j * 16);
    }
#pragma unroll
    for (int j = 0; j < 4; ++j) acc[j] = wmma_bf16(a1, b1[j], acc[j]);
  }
  int m = lane & 15, hh = lane >> 4;
#pragma unroll
  for (int j = 0; j < 4; ++j)
#pragma unroll
    for (int r = 0; r < 8; ++r) {
      int row = mtile * 16 + r + 8 * hh;            // ctx position 0..79
      int col = n0 + j * 16 + m;                    // inner 0..511
      if (ktype) dst[(size_t)row * INNER_ + col] = f2bf(acc[j][r]);
      else       dst[(size_t)col * LP_ + row]   = f2bf(acc[j][r]);
    }
}

// ---------- kernel 4: global max of scale * q_c @ k_c[0]^T ----------

__global__ void __launch_bounds__(256) sim0_max_kernel(const bf16_t* __restrict__ qbf,
                                                       const bf16_t* __restrict__ kbuf,
                                                       unsigned int* __restrict__ maxbuf) {
  int wave = threadIdx.x >> 5, lane = threadIdx.x & 31;
  int wid = blockIdx.x * 8 + wave;   // 0..2047
  int head = wid >> 8;
  int row0 = (wid & 255) * 16;
  const bf16_t* qh = qbf + head * DH_;
  const bf16_t* kh = kbuf + (size_t)1 * LP_ * INNER_ + head * DH_;  // k_c[0]
  v16bf aq0 = load_a(qh, INNER_, N_ + row0, 0);
  v16bf aq1 = load_a(qh, INNER_, N_ + row0, 32);
  float mx = -3.0e38f;
  int m = lane & 15;
  v16bf bk0 = load_b(kh, INNER_, 0, 0);
  v16bf bk1 = load_b(kh, INNER_, 32, 0);
#pragma unroll
  for (int nt = 0; nt < 5; ++nt) {
    v16bf bn0 = bk0, bn1 = bk1;
    if (nt + 1 < 5) {
      bn0 = load_b(kh, INNER_, 0, (nt + 1) * 16);
      bn1 = load_b(kh, INNER_, 32, (nt + 1) * 16);
    }
    v8f acc = vzero8();
    acc = wmma_bf16(aq0, bk0, acc);
    acc = wmma_bf16(aq1, bk1, acc);
    if (nt * 16 + m < L_) {
#pragma unroll
      for (int r = 0; r < 8; ++r) mx = fmaxf(mx, acc[r] * SCALE_);
    }
    bk0 = bn0; bk1 = bn1;
  }
#pragma unroll
  for (int off = 16; off >= 1; off >>= 1) mx = fmaxf(mx, __shfl_xor(mx, off, 32));
  if (lane == 0) {
    unsigned u = __builtin_bit_cast(unsigned, mx);
    unsigned enc = (u & 0x80000000u) ? ~u : (u | 0x80000000u);
    atomicMax(maxbuf, enc);
  }
}

// ---------- kernel 5: fused attention ----------

__global__ void __launch_bounds__(256) attention_kernel(const bf16_t* __restrict__ qbf,
                                                        const bf16_t* __restrict__ kbuf,
                                                        const bf16_t* __restrict__ vtbuf,
                                                        const float* __restrict__ attn_extra,
                                                        const unsigned int* __restrict__ maxbuf,
                                                        const int* __restrict__ tptr,
                                                        bf16_t* __restrict__ outbuf) {
  __shared__ __align__(16) bf16_t Plds[8][16 * LP_];
  int wave = threadIdx.x >> 5, lane = threadIdx.x & 31;
  int m = lane & 15, hh = lane >> 4;
  int bh = blockIdx.y;                 // 0..15
  bool cmode = bh >= H_;
  int head = bh & (H_ - 1);
  int row0 = blockIdx.x * 128 + wave * 16;
  int qrow0 = (cmode ? N_ : 0) + row0;
  const bf16_t* qh = qbf + head * DH_;
  v16bf aq0 = load_a(qh, INNER_, qrow0, 0);
  v16bf aq1 = load_a(qh, INNER_, qrow0, 32);
  float wmask = 0.0f;
  if (cmode) {
    unsigned u = *maxbuf;
    float smax = (u & 0x80000000u) ? __builtin_bit_cast(float, u & 0x7FFFFFFFu)
                                   : __builtin_bit_cast(float, ~u);
    float weight = (float)(*tptr) * (4.6f / 50.0f);
    wmask = 1.0f /*W_DOT*/ * weight * smax;
  }
  v8f oacc[4];
#pragma unroll
  for (int j = 0; j < 4; ++j) oacc[j] = vzero8();
  bf16_t* Pw = Plds[wave];
  int nctx = cmode ? C_ : 1;
  for (int c = 0; c < nctx; ++c) {
    int slot = cmode ? (1 + c) : 0;
    const bf16_t* kh = kbuf  + (size_t)slot * LP_ * INNER_ + head * DH_;
    const bf16_t* vh = vtbuf + (size_t)slot * LP_ * INNER_ + (size_t)head * DH_ * LP_;
    // sim = scale * q @ k^T   (16 x 80), fully unrolled + double-buffered
    v8f sim[5];
    v16bf bk0 = load_b(kh, INNER_, 0, 0);
    v16bf bk1 = load_b(kh, INNER_, 32, 0);
#pragma unroll
    for (int nt = 0; nt < 5; ++nt) {
      v16bf bn0 = bk0, bn1 = bk1;
      if (nt + 1 < 5) {
        bn0 = load_b(kh, INNER_, 0, (nt + 1) * 16);
        bn1 = load_b(kh, INNER_, 32, (nt + 1) * 16);
      }
      v8f acc = vzero8();
      acc = wmma_bf16(aq0, bk0, acc);
      acc = wmma_bf16(aq1, bk1, acc);
      sim[nt] = acc;
      bk0 = bn0; bk1 = bn1;
    }
    // scale, wmask*attn_extra (c==0 only), mask padded columns
#pragma unroll
    for (int nt = 0; nt < 5; ++nt) {
      int col = nt * 16 + m;
#pragma unroll
      for (int r = 0; r < 8; ++r) {
        float xv = sim[nt][r] * SCALE_;
        if (col < L_) {
          if (cmode && c == 0) {
            int rowg = row0 + r + 8 * hh;
            xv += wmask * attn_extra[((size_t)head * N_ + rowg) * L_ + col];
          }
        } else {
          xv = -1.0e30f;
        }
        sim[nt][r] = xv;
      }
    }
    // row softmax: row r+8*hh lives in the 16 lanes of this half-wave
#pragma unroll
    for (int r = 0; r < 8; ++r) {
      float pm = sim[0][r];
#pragma unroll
      for (int nt = 1; nt < 5; ++nt) pm = fmaxf(pm, sim[nt][r]);
#pragma unroll
      for (int off = 8; off >= 1; off >>= 1) pm = fmaxf(pm, __shfl_xor(pm, off, 16));
      float ps = 0.0f;
#pragma unroll
      for (int nt = 0; nt < 5; ++nt) {
        float e = __expf(sim[nt][r] - pm);
        sim[nt][r] = e;
        ps += e;
      }
#pragma unroll
      for (int off = 8; off >= 1; off >>= 1) ps += __shfl_xor(ps, off, 16);
      float inv = 1.0f / ps;
#pragma unroll
      for (int nt = 0; nt < 5; ++nt) sim[nt][r] *= inv;
    }
    // P: D-layout -> LDS -> A-layout (zero-fill K pad 80..95)
    unsigned int* P32 = (unsigned int*)Pw;
#pragma unroll
    for (int i = 0; i < (16 * LP_ / 2) / 32; ++i) P32[lane + 32 * i] = 0u;
    __syncthreads();
#pragma unroll
    for (int nt = 0; nt < 5; ++nt)
#pragma unroll
      for (int r = 0; r < 8; ++r)
        Pw[(r + 8 * hh) * LP_ + nt * 16 + m] = f2bf(sim[nt][r]);
    __syncthreads();
    // out += P(16x96) @ V(96x64); fully unrolled + double-buffered V
    v16bf bv_cur[4];
#pragma unroll
    for (int j = 0; j < 4; ++j) bv_cur[j] = load_b(vh, LP_, 0, j * 16);
#pragma unroll
    for (int ks = 0; ks < 3; ++ks) {
      v16bf bv_nxt[4];
#pragma unroll
      for (int j = 0; j < 4; ++j) bv_nxt[j] = bv_cur[j];
      if (ks + 1 < 3) {
#pragma unroll
        for (int j = 0; j < 4; ++j) bv_nxt[j] = load_b(vh, LP_, (ks + 1) * 32, j * 16);
      }
      v16bf a = load_a(Pw, LP_, 0, ks * 32);
#pragma unroll
      for (int j = 0; j < 4; ++j) oacc[j] = wmma_bf16(a, bv_cur[j], oacc[j]);
#pragma unroll
      for (int j = 0; j < 4; ++j) bv_cur[j] = bv_nxt[j];
    }
    __syncthreads();
  }
  float osc = cmode ? (1.0f / C_) : 1.0f;
#pragma unroll
  for (int j = 0; j < 4; ++j)
#pragma unroll
    for (int r = 0; r < 8; ++r)
      outbuf[(size_t)(qrow0 + r + 8 * hh) * INNER_ + head * DH_ + j * 16 + m] =
          f2bf(oacc[j][r] * osc);
}

// ---------- kernel 6: out = merged @ Wo + bo   (8192 x 320, K=512) ----------

__global__ void __launch_bounds__(256) final_kernel(const bf16_t* __restrict__ obuf,
                                                    const bf16_t* __restrict__ WoT,
                                                    const float* __restrict__ bo,
                                                    float* __restrict__ out) {
  int wave = threadIdx.x >> 5, lane = threadIdx.x & 31;
  int wid = blockIdx.x * 8 + wave;   // 0..2559
  int mtile = wid / 5;               // 0..511
  int ng = wid % 5;                  // 4 N-tiles each (320 = 5*64)
  int row0 = mtile * 16;
  int n0 = ng * 64;
  const int KS = INNER_ / 32;        // 16 (even)
  v8f acc[4];
#pragma unroll
  for (int j = 0; j < 4; ++j) acc[j] = vzero8();
  v16bf a0 = load_a(obuf, INNER_, row0, 0);
  v16bf b0[4];
#pragma unroll
  for (int j = 0; j < 4; ++j) b0[j] = load_b(WoT, INNER_, 0, n0 + j * 16);
  for (int ks = 0; ks < KS; ks += 2) {
    v16bf a1 = load_a(obuf, INNER_, row0, (ks + 1) * 32);
    v16bf b1[4];
#pragma unroll
    for (int j = 0; j < 4; ++j) b1[j] = load_b(WoT, INNER_, (ks + 1) * 32, n0 + j * 16);
#pragma unroll
    for (int j = 0; j < 4; ++j) acc[j] = wmma_bf16(a0, b0[j], acc[j]);
    if (ks + 2 < KS) {
      a0 = load_a(obuf, INNER_, row0, (ks + 2) * 32);
#pragma unroll
      for (int j = 0; j < 4; ++j) b0[j] = load_b(WoT, INNER_, (ks + 2) * 32, n0 + j * 16);
    }
#pragma unroll
    for (int j = 0; j < 4; ++j) acc[j] = wmma_bf16(a1, b1[j], acc[j]);
  }
  int m = lane & 15, hh = lane >> 4;
#pragma unroll
  for (int j = 0; j < 4; ++j)
#pragma unroll
    for (int r = 0; r < 8; ++r) {
      int col = n0 + j * 16 + m;
      out[(size_t)(row0 + r + 8 * hh) * DQ_ + col] = acc[j][r] + bo[col];
    }
}

// ---------- launch ----------

extern "C" void kernel_launch(void* const* d_in, const int* in_sizes, int n_in,
                              void* d_out, int out_size, void* d_ws, size_t ws_size,
                              hipStream_t stream) {
  (void)in_sizes; (void)n_in; (void)out_size; (void)ws_size;
  const float* x  = (const float*)d_in[0];
  const float* uc = (const float*)d_in[1];
  const float* ck = (const float*)d_in[2];
  const float* cv = (const float*)d_in[3];
  const float* ae = (const float*)d_in[4];
  const float* Wq = (const float*)d_in[5];
  const float* Wk = (const float*)d_in[6];
  const float* Wv = (const float*)d_in[7];
  const float* Wo = (const float*)d_in[8];
  const float* bo = (const float*)d_in[9];
  const int*   t  = (const int*)d_in[10];
  float* out = (float*)d_out;

  char* ws = (char*)d_ws;
  size_t off = 0;
  auto alloc = [&](size_t bytes) -> char* {
    char* p = ws + off;
    off = (off + bytes + 255) & ~(size_t)255;
    return p;
  };
  bf16_t* xbf   = (bf16_t*)alloc((size_t)ROWS_ * DQ_ * 2);
  bf16_t* WqT   = (bf16_t*)alloc((size_t)INNER_ * DQ_ * 2);
  bf16_t* WkT   = (bf16_t*)alloc((size_t)INNER_ * DC_ * 2);
  bf16_t* WvT   = (bf16_t*)alloc((size_t)INNER_ * DC_ * 2);
  bf16_t* WoT   = (bf16_t*)alloc((size_t)DQ_ * INNER_ * 2);
  bf16_t* qbf   = (bf16_t*)alloc((size_t)ROWS_ * INNER_ * 2);
  bf16_t* kbuf  = (bf16_t*)alloc((size_t)5 * LP_ * INNER_ * 2);
  bf16_t* vtbuf = (bf16_t*)alloc((size_t)5 * LP_ * INNER_ * 2);
  bf16_t* obuf  = (bf16_t*)alloc((size_t)ROWS_ * INNER_ * 2);
  unsigned int* maxbuf = (unsigned int*)alloc(256);

  prep_kernel<<<2048, 256, 0, stream>>>(x, Wq, Wk, Wv, Wo, xbf, WqT, WkT, WvT, WoT, maxbuf);
  proj_q_kernel<<<512, 256, 0, stream>>>(xbf, WqT, qbf);
  proj_ctx_kernel<<<50, 256, 0, stream>>>(uc, ck, cv, WkT, WvT, kbuf, vtbuf);
  sim0_max_kernel<<<256, 256, 0, stream>>>(qbf, kbuf, maxbuf);
  attention_kernel<<<dim3(32, 16), 256, 0, stream>>>(qbf, kbuf, vtbuf, ae, maxbuf, t, obuf);
  final_kernel<<<320, 256, 0, stream>>>(obuf, WoT, bo, out);
}
